// MedVidNet_76330158784766
// MI455X (gfx1250) — matrix operand order repaired and checked by reading, not compiled, not run on hardware
//
#include <hip/hip_runtime.h>
#include <math.h>

#define T_DIM 65536
#define S_DIM 64
#define F_DIM 2048
#define H_DIM 16
#define D_DIM 128
#define L_DIM 64
#define INV_SCALE 0.08838834764831845f /* 1/sqrt(128) */

typedef __attribute__((ext_vector_type(16))) __bf16 v16bf;
typedef __attribute__((ext_vector_type(8)))  float  v8f;

// ---------------------------------------------------------------------------
// Kernel 1: lamb[t][h] = sum_l tanh( sum_d hq[t,h,d] * V[h,l,d] ) * w[h,l]
// One wave = 16 frames x 1 head, all 64 L columns via 4 N-tiles.
// bf16 WMMA (v_wmma_f32_16x16x32_bf16), f32 accumulate. 16 WMMA per wave.
// ---------------------------------------------------------------------------
__global__ __launch_bounds__(256) void lamb_kernel(
    const float* __restrict__ videos,   // [T, F]
    const float* __restrict__ Vmat,     // [H, L, D]
    const float* __restrict__ w,        // [H, L]
    float* __restrict__ lamb)           // [T, H]
{
    const int h     = blockIdx.y;
    const int lane  = threadIdx.x & 31;
    const int wave  = threadIdx.x >> 5;        // 0..7
    const int col   = lane & 15;               // N column / M row selector
    const int khalf = lane >> 4;               // 0 or 1
    const int tbase = blockIdx.x * 128 + wave * 16;

    // A rows: lane (0..15 | 16..31) holds row M = lane&15
    const float* arow = videos + (size_t)(tbase + col) * F_DIM + (size_t)h * D_DIM;

    v8f c[4];
    #pragma unroll
    for (int nt = 0; nt < 4; ++nt) c[nt] = (v8f){0.f,0.f,0.f,0.f,0.f,0.f,0.f,0.f};

    #pragma unroll
    for (int kc = 0; kc < 4; ++kc) {
        const int kk = kc * 32;

        // --- A fragment, 16x32 bf16 (ISA 7.12.2 layout):
        // lanes 0-15 hold K = {kk+0..7, kk+16..23}; lanes 16-31 the +8 halves.
        const float* ap = arow + kk + khalf * 8;
        float4 x0 = *(const float4*)(ap + 0);
        float4 x1 = *(const float4*)(ap + 4);
        float4 x2 = *(const float4*)(ap + 16);
        float4 x3 = *(const float4*)(ap + 20);
        v16bf a;
        a[0]  = (__bf16)x0.x; a[1]  = (__bf16)x0.y; a[2]  = (__bf16)x0.z; a[3]  = (__bf16)x0.w;
        a[4]  = (__bf16)x1.x; a[5]  = (__bf16)x1.y; a[6]  = (__bf16)x1.z; a[7]  = (__bf16)x1.w;
        a[8]  = (__bf16)x2.x; a[9]  = (__bf16)x2.y; a[10] = (__bf16)x2.z; a[11] = (__bf16)x2.w;
        a[12] = (__bf16)x3.x; a[13] = (__bf16)x3.y; a[14] = (__bf16)x3.z; a[15] = (__bf16)x3.w;

        #pragma unroll
        for (int nt = 0; nt < 4; ++nt) {
            // --- B fragment, 32x16 bf16: lane = column L, same K order as A
            // (lanes 0-15: K={kk+0..7, kk+16..23}; lanes 16-31 the +8 halves).
            // B[d][l] = V[h][l][d]  (V row-major over d => contiguous load).
            const int l = nt * 16 + col;
            const float* bp = Vmat + ((size_t)h * L_DIM + l) * D_DIM + kk + khalf * 8;
            float4 y0 = *(const float4*)(bp + 0);
            float4 y1 = *(const float4*)(bp + 4);
            float4 y2 = *(const float4*)(bp + 16);
            float4 y3 = *(const float4*)(bp + 20);
            v16bf b;
            b[0]  = (__bf16)y0.x; b[1]  = (__bf16)y0.y; b[2]  = (__bf16)y0.z; b[3]  = (__bf16)y0.w;
            b[4]  = (__bf16)y1.x; b[5]  = (__bf16)y1.y; b[6]  = (__bf16)y1.z; b[7]  = (__bf16)y1.w;
            b[8]  = (__bf16)y2.x; b[9]  = (__bf16)y2.y; b[10] = (__bf16)y2.z; b[11] = (__bf16)y2.w;
            b[12] = (__bf16)y3.x; b[13] = (__bf16)y3.y; b[14] = (__bf16)y3.z; b[15] = (__bf16)y3.w;

            c[nt] = __builtin_amdgcn_wmma_f32_16x16x32_bf16(
                /*neg_a=*/false, a, /*neg_b=*/false, b,
                /*c_mod=*/(short)0, c[nt], /*reuse_a=*/false, /*reuse_b=*/false);
        }
    }

    // C/D layout: lane's column n = (lane&15)+nt*16, rows m = v + 8*(lane>>4).
    // lamb partial per row: sum over this lane's 4 columns of tanh(alpha)*w.
    float wl[4];
    #pragma unroll
    for (int nt = 0; nt < 4; ++nt) wl[nt] = w[h * L_DIM + nt * 16 + col];

    #pragma unroll
    for (int v = 0; v < 8; ++v) {
        float p = 0.f;
        #pragma unroll
        for (int nt = 0; nt < 4; ++nt) p += tanhf(c[nt][v]) * wl[nt];
        // reduce the 16 columns held by this half of the wave
        #pragma unroll
        for (int mask = 1; mask < 16; mask <<= 1) p += __shfl_xor(p, mask, 32);
        if (col == 0) {
            lamb[(size_t)(tbase + khalf * 8 + v) * H_DIM + h] = p;
        }
    }
}

// ---------------------------------------------------------------------------
// Block reduction helper (max or sum) for 256-thread blocks.
// ---------------------------------------------------------------------------
__device__ __forceinline__ float block_reduce(float v, float* sm, bool is_max)
{
    #pragma unroll
    for (int mask = 16; mask; mask >>= 1) {
        float o = __shfl_xor(v, mask, 32);
        v = is_max ? fmaxf(v, o) : (v + o);
    }
    const int lane = threadIdx.x & 31;
    const int wave = threadIdx.x >> 5;
    __syncthreads();
    if (lane == 0) sm[wave] = v;
    __syncthreads();
    const int nw = (blockDim.x + 31) >> 5;
    if (wave == 0) {
        v = (lane < nw) ? sm[lane] : (is_max ? -INFINITY : 0.f);
        #pragma unroll
        for (int mask = 16; mask; mask >>= 1) {
            float o = __shfl_xor(v, mask, 32);
            v = is_max ? fmaxf(v, o) : (v + o);
        }
        if (lane == 0) sm[0] = v;
    }
    __syncthreads();
    return sm[0];
}

__device__ __forceinline__ int lower_bound_seg(const int* __restrict__ seg, int key)
{
    int lo = 0, hi = T_DIM;
    while (lo < hi) {
        int mid = (lo + hi) >> 1;
        if (seg[mid] < key) lo = mid + 1; else hi = mid;
    }
    return lo;
}

// ---------------------------------------------------------------------------
// Kernel 2: per-(study, head) softmax stats: max and 1/denom.
// ---------------------------------------------------------------------------
__global__ __launch_bounds__(256) void seg_stats_kernel(
    const float* __restrict__ lamb,     // [T, H]
    const int*   __restrict__ seg,      // [T], sorted
    float* __restrict__ smax,           // [S, H]
    float* __restrict__ sinv)           // [S, H]
{
    __shared__ float sm[8];
    __shared__ int bounds[2];
    const int s = blockIdx.x;
    const int h = blockIdx.y;
    if (threadIdx.x == 0) {
        bounds[0] = lower_bound_seg(seg, s);
        bounds[1] = lower_bound_seg(seg, s + 1);
    }
    __syncthreads();
    const int sb = bounds[0], se = bounds[1];

    float mx = -INFINITY;
    for (int t = sb + threadIdx.x; t < se; t += blockDim.x)
        mx = fmaxf(mx, lamb[(size_t)t * H_DIM + h]);
    mx = block_reduce(mx, sm, true);

    float sum = 0.f;
    for (int t = sb + threadIdx.x; t < se; t += blockDim.x)
        sum += __expf(lamb[(size_t)t * H_DIM + h] - mx);
    sum = block_reduce(sum, sm, false);

    if (threadIdx.x == 0) {
        smax[s * H_DIM + h] = mx;
        sinv[s * H_DIM + h] = (sum > 0.f) ? (1.f / sum) : 0.f;
    }
}

// ---------------------------------------------------------------------------
// Kernel 3: h_study[s,h,d] += sum_t attn(t,h) * hq[t,h,d] / SCALE.
// Coalesced: 128 threads = d, 512B per frame. 4 t-chunks per segment,
// f32 global atomics merge the partials.
// ---------------------------------------------------------------------------
__global__ __launch_bounds__(128) void seg_wsum_kernel(
    const float* __restrict__ videos,   // [T, F]
    const float* __restrict__ lamb,     // [T, H]
    const int*   __restrict__ seg,      // [T]
    const float* __restrict__ smax,     // [S, H]
    const float* __restrict__ sinv,     // [S, H]
    float* __restrict__ hstudy)         // [S, F] (pre-zeroed)
{
    __shared__ int bounds[2];
    const int s  = blockIdx.x;
    const int h  = blockIdx.y;
    const int ch = blockIdx.z;          // 0..3
    if (threadIdx.x == 0) {
        bounds[0] = lower_bound_seg(seg, s);
        bounds[1] = lower_bound_seg(seg, s + 1);
    }
    __syncthreads();
    const int sb = bounds[0], se = bounds[1];
    const int len = se - sb;
    const int per = (len + 3) >> 2;
    const int cs = sb + ch * per;
    const int ce = min(se, cs + per);

    const float mx  = smax[s * H_DIM + h];
    const float inv = sinv[s * H_DIM + h] * INV_SCALE;
    const int d = threadIdx.x;

    float acc = 0.f;
    for (int t = cs; t < ce; ++t) {
        float wgt = __expf(lamb[(size_t)t * H_DIM + h] - mx);
        acc += wgt * videos[(size_t)t * F_DIM + h * D_DIM + d];
    }
    acc *= inv;
    atomicAdd(&hstudy[(size_t)s * F_DIM + h * D_DIM + d], acc);
}

// ---------------------------------------------------------------------------
// Kernel 4: output[s] = h_study[s,:] . W_out + b_out
// ---------------------------------------------------------------------------
__global__ __launch_bounds__(256) void out_kernel(
    const float* __restrict__ hstudy,   // [S, F]
    const float* __restrict__ Wout,     // [F]
    const float* __restrict__ bout,     // [1]
    float* __restrict__ out)            // [S]
{
    __shared__ float sm[8];
    const int s = blockIdx.x;
    float sum = 0.f;
    for (int f = threadIdx.x; f < F_DIM; f += blockDim.x)
        sum += hstudy[(size_t)s * F_DIM + f] * Wout[f];
    sum = block_reduce(sum, sm, false);
    if (threadIdx.x == 0) out[s] = sum + bout[0];
}

// ---------------------------------------------------------------------------
extern "C" void kernel_launch(void* const* d_in, const int* in_sizes, int n_in,
                              void* d_out, int out_size, void* d_ws, size_t ws_size,
                              hipStream_t stream)
{
    const float* videos = (const float*)d_in[0];   // [T, F]
    const int*   seg    = (const int*)  d_in[1];   // [T]
    const float* Vmat   = (const float*)d_in[2];   // [H, L, D]
    const float* w      = (const float*)d_in[3];   // [H, L]
    const float* Wout   = (const float*)d_in[4];   // [F, 1]
    const float* bout   = (const float*)d_in[5];   // [1]
    float* out = (float*)d_out;                    // [S, 1]

    char* ws = (char*)d_ws;
    float* lamb   = (float*)ws;                                       // T*H
    float* smax   = (float*)(ws + (size_t)T_DIM * H_DIM * sizeof(float));
    float* sinv   = smax + (size_t)S_DIM * H_DIM;
    float* hstudy = sinv + (size_t)S_DIM * H_DIM;                     // S*F

    hipMemsetAsync(hstudy, 0, (size_t)S_DIM * F_DIM * sizeof(float), stream);

    lamb_kernel<<<dim3(T_DIM / 128, H_DIM), 256, 0, stream>>>(videos, Vmat, w, lamb);
    seg_stats_kernel<<<dim3(S_DIM, H_DIM), 256, 0, stream>>>(lamb, seg, smax, sinv);
    seg_wsum_kernel<<<dim3(S_DIM, H_DIM, 4), 128, 0, stream>>>(videos, lamb, seg,
                                                               smax, sinv, hstudy);
    out_kernel<<<S_DIM, 256, 0, stream>>>(hstudy, Wout, bout, out);
}
